// PointNet2_53060025974803
// MI455X (gfx1250) — compile-verified, matrix operand order checked by reading, not run on hardware
//
#include <hip/hip_runtime.h>
#include <hip/hip_bf16.h>

typedef __attribute__((ext_vector_type(16))) _Float16 v16h;
typedef __attribute__((ext_vector_type(8)))  _Float16 v8h;
typedef __attribute__((ext_vector_type(8)))  float    v8f;

// ---------------------------------------------------------------------------
// Farthest point sampling: one workgroup per batch element, min-dist in LDS.
// ---------------------------------------------------------------------------
__global__ void fps_k(const float* __restrict__ co, int n, int ns,
                      int* __restrict__ idx_out, float* __restrict__ co_out) {
  __shared__ float mind[4096];
  __shared__ float redv[256];
  __shared__ int   redi[256];
  __shared__ int   far_s;
  const int b = blockIdx.x;
  const int t = threadIdx.x;
  const float* cob = co + (size_t)b * n * 3;
  for (int i = t; i < n; i += blockDim.x) mind[i] = 1e10f;
  if (t == 0) far_s = 0;
  __syncthreads();
  for (int s = 0; s < ns; ++s) {
    const int far = far_s;
    const float cx = cob[far * 3 + 0];
    const float cy = cob[far * 3 + 1];
    const float cz = cob[far * 3 + 2];
    if (t == 0) {
      idx_out[b * ns + s] = far;
      co_out[((size_t)b * ns + s) * 3 + 0] = cx;
      co_out[((size_t)b * ns + s) * 3 + 1] = cy;
      co_out[((size_t)b * ns + s) * 3 + 2] = cz;
    }
    float bestv = -1.0f; int besti = 0;
    for (int i = t; i < n; i += blockDim.x) {
      const float dx = cob[i * 3 + 0] - cx;
      const float dy = cob[i * 3 + 1] - cy;
      const float dz = cob[i * 3 + 2] - cz;
      const float d = dx * dx + dy * dy + dz * dz;
      const float md = fminf(mind[i], d);
      mind[i] = md;
      if (md > bestv) { bestv = md; besti = i; }
    }
    redv[t] = bestv; redi[t] = besti;
    __syncthreads();
    for (int off = blockDim.x >> 1; off > 0; off >>= 1) {
      if (t < off) {
        const float ov = redv[t + off]; const int oi = redi[t + off];
        if (ov > redv[t] || (ov == redv[t] && oi < redi[t])) { redv[t] = ov; redi[t] = oi; }
      }
      __syncthreads();
    }
    if (t == 0) far_s = redi[0];
    __syncthreads();
  }
}

// ---------------------------------------------------------------------------
// Ball query: first K in-ball indices (ascending), padded with first valid.
// ---------------------------------------------------------------------------
__global__ void ball_query_k(const float* __restrict__ co,
                             const float* __restrict__ cent,
                             int B, int n, int S, int K, float r2,
                             int* __restrict__ gidx) {
  const int gs = blockIdx.x * blockDim.x + threadIdx.x;
  if (gs >= B * S) return;
  const int b = gs / S;
  const float* cb = co + (size_t)b * n * 3;
  const float cx = cent[(size_t)gs * 3 + 0];
  const float cy = cent[(size_t)gs * 3 + 1];
  const float cz = cent[(size_t)gs * 3 + 2];
  int* out = gidx + (size_t)gs * K;
  int cnt = 0, first = 0;
  for (int i = 0; i < n && cnt < K; ++i) {
    const float dx = cb[i * 3 + 0] - cx;
    const float dy = cb[i * 3 + 1] - cy;
    const float dz = cb[i * 3 + 2] - cz;
    if (dx * dx + dy * dy + dz * dz <= r2) {
      if (cnt == 0) first = i;
      out[cnt++] = i;
    }
  }
  for (; cnt < K; ++cnt) out[cnt] = first;
}

// ---------------------------------------------------------------------------
// Grouped GEMM-A builder: X[b,s,k, 0:3+Cf] = [g_co - cent, fea], f16, rows
// padded with zeros up to Kp (multiple of 32).
// ---------------------------------------------------------------------------
__global__ void group_h_k(const float* __restrict__ co, const float* __restrict__ fea,
                          int B, int n, int Cf, const float* __restrict__ cent,
                          const int* __restrict__ gidx, int S, int K, int Kp,
                          _Float16* __restrict__ X) {
  const long long tid = (long long)blockIdx.x * blockDim.x + threadIdx.x;
  const long long tot = (long long)B * S * K;
  if (tid >= tot) return;
  const long long bs = tid / K;
  const int b = (int)(bs / S);
  const int gi = gidx[tid];
  const float* p = co + ((size_t)b * n + gi) * 3;
  _Float16* x = X + (size_t)tid * Kp;
  x[0] = (_Float16)(p[0] - cent[(size_t)bs * 3 + 0]);
  x[1] = (_Float16)(p[1] - cent[(size_t)bs * 3 + 1]);
  x[2] = (_Float16)(p[2] - cent[(size_t)bs * 3 + 2]);
  const float* f = fea + ((size_t)b * n + gi) * Cf;
  for (int c = 0; c < Cf; ++c) x[3 + c] = (_Float16)f[c];
  for (int c = 3 + Cf; c < Kp; ++c) x[c] = (_Float16)0.0f;
}

// SA3 group: Xh[b,s,c] = (f16) fea[b, idx[b,s], c]   (C multiple of 32)
__global__ void gather_fea_h_k(const float* __restrict__ fea, const int* __restrict__ idx,
                               int B, int n, int S, int C, _Float16* __restrict__ out) {
  const long long tid = (long long)blockIdx.x * blockDim.x + threadIdx.x;
  const long long tot = (long long)B * S * C;
  if (tid >= tot) return;
  const int c = (int)(tid % C);
  const long long bs = tid / C;
  const int b = (int)(bs / S);
  const int s = (int)(bs % S);
  out[tid] = (_Float16)fea[((size_t)b * n + idx[b * S + s]) * C + c];
}

// Weight repack: Wh[Np,Kp] f16 zero-padded from W[N,K] f32.
__global__ void repack_w_k(const float* __restrict__ W, _Float16* __restrict__ Wh,
                           int N, int K, int Np, int Kp) {
  const long long tid = (long long)blockIdx.x * blockDim.x + threadIdx.x;
  if (tid >= (long long)Np * Kp) return;
  const int nr = (int)(tid / Kp);
  const int k = (int)(tid % Kp);
  Wh[tid] = (nr < N && k < K) ? (_Float16)W[(size_t)nr * K + k] : (_Float16)0.0f;
}

// ---------------------------------------------------------------------------
// WMMA GEMM: Y[M,N] = Xh[M,Kp] @ Wh[Np,Kp]^T + bias[N].  f16 operands (pre-
// padded, no bounds checks in the k-loop), f32 accumulate. 8 waves / block,
// one 16x16 tile per wave. Inner loop: 4x b128 loads + 1 v_wmma.
// ---------------------------------------------------------------------------
__global__ void gemm_bias_wmma(const _Float16* __restrict__ Xh, const _Float16* __restrict__ Wh,
                               const float* __restrict__ bias, float* __restrict__ Y,
                               int Kp, int N, int tilesM, int tilesN) {
  const int wave = threadIdx.x >> 5;
  const int tile = blockIdx.x * 8 + wave;
  if (tile >= tilesM * tilesN) return;          // wave-uniform: EXEC stays full
  const int tM = tile % tilesM;
  const int tN = tile / tilesM;
  const int lane = threadIdx.x & 31;
  const int lane16 = lane & 15;
  const int hi8 = (lane >> 4) << 3;
  const _Float16* arow = Xh + (size_t)(tM * 16 + lane16) * Kp + hi8;
  const _Float16* brow = Wh + (size_t)(tN * 16 + lane16) * Kp + hi8;
  v8f acc = {};
  for (int k0 = 0; k0 < Kp; k0 += 32) {
    const v8h a0 = *(const v8h*)(arow + k0);
    const v8h a1 = *(const v8h*)(arow + k0 + 16);
    const v8h b0 = *(const v8h*)(brow + k0);
    const v8h b1 = *(const v8h*)(brow + k0 + 16);
    const v16h a = __builtin_shufflevector(a0, a1, 0, 1, 2, 3, 4, 5, 6, 7,
                                           8, 9, 10, 11, 12, 13, 14, 15);
    const v16h b = __builtin_shufflevector(b0, b1, 0, 1, 2, 3, 4, 5, 6, 7,
                                           8, 9, 10, 11, 12, 13, 14, 15);
    acc = __builtin_amdgcn_wmma_f32_16x16x32_f16(
        false, a, false, b, (short)0, acc, false, false);
  }
  const int nc = tN * 16 + lane16;
  if (nc < N) {
    const float bv = bias[nc];
#pragma unroll
    for (int r = 0; r < 8; ++r) {
      const int mr = tM * 16 + r + hi8;
      Y[(size_t)mr * N + nc] = acc[r] + bv;
    }
  }
}

// Per-channel sum / sumsq over M rows (one block per channel).
__global__ void bn_stats_k(const float* __restrict__ Y, long long M, int N,
                           float* __restrict__ stats) {
  const int n = blockIdx.x;
  float s = 0.0f, sq = 0.0f;
  for (long long m = threadIdx.x; m < M; m += blockDim.x) {
    const float v = Y[(size_t)m * N + n];
    s += v; sq += v * v;
  }
  __shared__ float sh[256], shq[256];
  sh[threadIdx.x] = s; shq[threadIdx.x] = sq;
  __syncthreads();
  for (int off = blockDim.x >> 1; off > 0; off >>= 1) {
    if (threadIdx.x < off) { sh[threadIdx.x] += sh[threadIdx.x + off]; shq[threadIdx.x] += shq[threadIdx.x + off]; }
    __syncthreads();
  }
  if (threadIdx.x == 0) { stats[n] = sh[0]; stats[N + n] = shq[0]; }
}

// BN+ReLU: update Y f32 in place AND emit next layer's f16 A-operand.
// (N is a multiple of 32 for every BN layer, so Xnext stride == N.)
__global__ void bn_apply_relu_k(float* __restrict__ Y, _Float16* __restrict__ Xnext,
                                long long M, int N,
                                const float* __restrict__ stats,
                                const float* __restrict__ g,
                                const float* __restrict__ beta) {
  const long long i = (long long)blockIdx.x * blockDim.x + threadIdx.x;
  if (i >= M * N) return;
  const int n = (int)(i % N);
  const float invM = 1.0f / (float)M;
  const float mean = stats[n] * invM;
  const float var  = fmaxf(stats[N + n] * invM - mean * mean, 0.0f);
  const float a = g[n] * rsqrtf(var + 1e-5f);
  float v = (Y[i] - mean) * a + beta[n];
  v = v > 0.0f ? v : 0.0f;
  Y[i] = v;
  Xnext[i] = (_Float16)v;
}

// max over group dim: X[rows, K, C] -> out[rows, C]  (f32)
__global__ void maxpool_k(const float* __restrict__ X, float* __restrict__ out,
                          long long rows, int K, int C) {
  const long long tid = (long long)blockIdx.x * blockDim.x + threadIdx.x;
  if (tid >= rows * C) return;
  const long long r = tid / C;
  const int c = (int)(tid % C);
  float v = -3.4e38f;
  for (int k = 0; k < K; ++k) v = fmaxf(v, X[((size_t)r * K + k) * C + c]);
  out[tid] = v;
}

// Concat helper: dst_f16[r, coff:coff+Cs] = (f16) src_f32[r, :]
__global__ void copy_cols_h_k(const float* __restrict__ src, _Float16* __restrict__ dst,
                              long long R, int Cs, int Ctot, int coff) {
  const long long i = (long long)blockIdx.x * blockDim.x + threadIdx.x;
  if (i >= R * Cs) return;
  const long long r = i / Cs;
  const int c = (int)(i % Cs);
  dst[(size_t)r * Ctot + coff + c] = (_Float16)src[i];
}

// ---------------------------------------------------------------------------
// FP interpolation: 3-NN inverse-distance; distances on the fly; writes f16
// into concat buffer at column offset coff.
// ---------------------------------------------------------------------------
__global__ void fp_interp_h_k(const float* __restrict__ co_s, const float* __restrict__ co_b,
                              const float* __restrict__ fea_b, int B, int n, int m, int C,
                              _Float16* __restrict__ out, int Ctot, int coff) {
  const long long gs = (long long)blockIdx.x * blockDim.x + threadIdx.x;
  if (gs >= (long long)B * n) return;
  const int b = (int)(gs / n);
  const float px = co_s[(size_t)gs * 3 + 0];
  const float py = co_s[(size_t)gs * 3 + 1];
  const float pz = co_s[(size_t)gs * 3 + 2];
  const float* cb = co_b + (size_t)b * m * 3;
  float d0 = 3e38f, d1 = 3e38f, d2 = 3e38f;
  int i0 = 0, i1 = 0, i2 = 0;
  for (int j = 0; j < m; ++j) {
    const float dx = cb[j * 3 + 0] - px;
    const float dy = cb[j * 3 + 1] - py;
    const float dz = cb[j * 3 + 2] - pz;
    const float d = dx * dx + dy * dy + dz * dz;
    if (d < d0)      { d2 = d1; i2 = i1; d1 = d0; i1 = i0; d0 = d; i0 = j; }
    else if (d < d1) { d2 = d1; i2 = i1; d1 = d;  i1 = j; }
    else if (d < d2) { d2 = d;  i2 = j; }
  }
  const float w0 = 1.0f / fmaxf(d0, 1e-10f);
  const float w1 = 1.0f / fmaxf(d1, 1e-10f);
  const float w2 = 1.0f / fmaxf(d2, 1e-10f);
  const float wsum = w0 + w1 + w2;
  const bool isbig = (d0 <= 1e-8f);
  const float* f0 = fea_b + ((size_t)b * m + i0) * C;
  const float* f1 = fea_b + ((size_t)b * m + i1) * C;
  const float* f2 = fea_b + ((size_t)b * m + i2) * C;
  _Float16* o = out + (size_t)gs * Ctot + coff;
  for (int c = 0; c < C; ++c) {
    const float v = isbig ? f0[c] : (w0 * f0[c] + w1 * f1[c] + w2 * f2[c]) / wsum;
    o[c] = (_Float16)v;
  }
}

// Softmax over the POINT axis of [b,n,C]; one block per (b, channel).
__global__ void softmax_points_k(float* __restrict__ out, int n, int C) {
  float* p = out + (size_t)blockIdx.x * n * C + blockIdx.y;
  __shared__ float sh[256];
  float mx = -3.4e38f;
  for (int i = threadIdx.x; i < n; i += blockDim.x) mx = fmaxf(mx, p[(size_t)i * C]);
  sh[threadIdx.x] = mx; __syncthreads();
  for (int off = blockDim.x >> 1; off > 0; off >>= 1) {
    if (threadIdx.x < off) sh[threadIdx.x] = fmaxf(sh[threadIdx.x], sh[threadIdx.x + off]);
    __syncthreads();
  }
  mx = sh[0]; __syncthreads();
  float s = 0.0f;
  for (int i = threadIdx.x; i < n; i += blockDim.x) s += expf(p[(size_t)i * C] - mx);
  sh[threadIdx.x] = s; __syncthreads();
  for (int off = blockDim.x >> 1; off > 0; off >>= 1) {
    if (threadIdx.x < off) sh[threadIdx.x] += sh[threadIdx.x + off];
    __syncthreads();
  }
  s = sh[0];
  for (int i = threadIdx.x; i < n; i += blockDim.x) {
    const size_t k = (size_t)i * C;
    p[k] = expf(p[k] - mx) / s;
  }
}

// ---------------------------------------------------------------------------
// Host orchestration
// ---------------------------------------------------------------------------
struct LayerP {
  const float *W, *b, *g, *beta;
  _Float16* Wh;
  int cin, cout, Kp, Np;
};

extern "C" void kernel_launch(void* const* d_in, const int* in_sizes, int n_in,
                              void* d_out, int out_size, void* d_ws, size_t ws_size,
                              hipStream_t stream) {
  (void)out_size; (void)ws_size;
  const float* pts = (const float*)d_in[0];

  LayerP sa1[3], sa2[3], sa3[3], fp3[2], fp2[2], fp1[4], head;
  int pi = 1;
  // Flatten convention detect: insertion order -> first W is sa1 (6*64=384);
  // jax tree_flatten (sorted keys) -> first W is fp1 (128*128).
  const bool sorted = (n_in > 1 && in_sizes[1] == 128 * 128);
  auto take = [&](LayerP& L, int cin, int cout) {
    L.cin = cin; L.cout = cout;
    L.W = (const float*)d_in[pi++];
    L.b = (const float*)d_in[pi++];
    if (sorted) { L.beta = (const float*)d_in[pi++]; L.g = (const float*)d_in[pi++]; }
    else        { L.g = (const float*)d_in[pi++]; L.beta = (const float*)d_in[pi++]; }
  };
  if (!sorted) {
    take(sa1[0], 6, 64);    take(sa1[1], 64, 64);   take(sa1[2], 64, 128);
    take(sa2[0], 131, 128); take(sa2[1], 128, 128); take(sa2[2], 128, 256);
    take(sa3[0], 256, 256); take(sa3[1], 256, 512); take(sa3[2], 512, 1024);
    take(fp3[0], 1280, 256); take(fp3[1], 256, 256);
    take(fp2[0], 384, 256);  take(fp2[1], 256, 128);
    take(fp1[0], 128, 128); take(fp1[1], 128, 128); take(fp1[2], 128, 128); take(fp1[3], 128, 128);
    take(head, 128, 2);
  } else {
    take(fp1[0], 128, 128); take(fp1[1], 128, 128); take(fp1[2], 128, 128); take(fp1[3], 128, 128);
    take(fp2[0], 384, 256);  take(fp2[1], 256, 128);
    take(fp3[0], 1280, 256); take(fp3[1], 256, 256);
    take(head, 128, 2);
    take(sa1[0], 6, 64);    take(sa1[1], 64, 64);   take(sa1[2], 64, 128);
    take(sa2[0], 131, 128); take(sa2[1], 128, 128); take(sa2[2], 128, 256);
    take(sa3[0], 256, 256); take(sa3[1], 256, 512); take(sa3[2], 512, 1024);
  }

  // Workspace bump allocator
  char* wp = (char*)d_ws;
  auto alloc = [&](size_t bytes) -> void* {
    void* r = (void*)wp;
    wp += (bytes + 255) & ~(size_t)255;
    return r;
  };
  const size_t MAXELEM = (size_t)67108864;            // 524288 x 128
  float*    bufY = (float*)alloc(MAXELEM * 4);        // f32 GEMM output / BN tensor
  _Float16* H0   = (_Float16*)alloc(MAXELEM * 2);     // f16 A-operand ping
  _Float16* H1   = (_Float16*)alloc(MAXELEM * 2);     // f16 A-operand pong
  float* fea1  = (float*)alloc((size_t)32 * 512 * 128 * 4);
  float* fea2  = (float*)alloc((size_t)32 * 128 * 256 * 4);
  float* fea3  = (float*)alloc((size_t)32 * 64 * 1024 * 4);
  float* co1   = (float*)alloc((size_t)32 * 512 * 3 * 4);
  float* co2   = (float*)alloc((size_t)32 * 128 * 3 * 4);
  float* co3   = (float*)alloc((size_t)32 * 64 * 3 * 4);
  int*   idx1  = (int*)alloc((size_t)32 * 512 * 4);
  int*   idx2  = (int*)alloc((size_t)32 * 128 * 4);
  int*   idx3  = (int*)alloc((size_t)32 * 64 * 4);
  int*   gidx1 = (int*)alloc((size_t)32 * 512 * 32 * 4);
  int*   gidx2 = (int*)alloc((size_t)32 * 128 * 64 * 4);
  float* stats = (float*)alloc((size_t)2 * 1024 * 4);

  const int Bc = 32;
  auto cdiv = [](long long a, long long b) -> unsigned { return (unsigned)((a + b - 1) / b); };

  // Repack all weights to padded f16
  LayerP* all[] = { &sa1[0], &sa1[1], &sa1[2], &sa2[0], &sa2[1], &sa2[2],
                    &sa3[0], &sa3[1], &sa3[2], &fp3[0], &fp3[1], &fp2[0], &fp2[1],
                    &fp1[0], &fp1[1], &fp1[2], &fp1[3], &head };
  for (int li = 0; li < 18; ++li) {
    LayerP& L = *all[li];
    L.Kp = (L.cin + 31) & ~31;
    L.Np = (L.cout + 15) & ~15;
    L.Wh = (_Float16*)alloc((size_t)L.Np * L.Kp * 2);
    const long long tot = (long long)L.Np * L.Kp;
    repack_w_k<<<cdiv(tot, 256), 256, 0, stream>>>(L.W, L.Wh, L.cout, L.cin, L.Np, L.Kp);
  }

  auto run_gemm = [&](const _Float16* Xh, float* Y, int M, const LayerP& L) {
    const int tilesM = M / 16, tilesN = L.Np / 16;
    gemm_bias_wmma<<<cdiv((long long)tilesM * tilesN, 8), 256, 0, stream>>>(
        Xh, L.Wh, L.b, Y, L.Kp, L.cout, tilesM, tilesN);
  };
  auto run_layer = [&](const _Float16* Xh, float* Y, _Float16* Xn, int M, const LayerP& L) {
    run_gemm(Xh, Y, M, L);
    bn_stats_k<<<L.cout, 256, 0, stream>>>(Y, (long long)M, L.cout, stats);
    const long long tot = (long long)M * L.cout;
    bn_apply_relu_k<<<cdiv(tot, 256), 256, 0, stream>>>(Y, Xn, (long long)M, L.cout, stats, L.g, L.beta);
  };

  // ---------------- SA1: 4096 -> 512, r=0.2, K=32, [6,64,64,128] -----------
  fps_k<<<Bc, 256, 0, stream>>>(pts, 4096, 512, idx1, co1);
  ball_query_k<<<cdiv((long long)Bc * 512, 256), 256, 0, stream>>>(pts, co1, Bc, 4096, 512, 32, 0.04f, gidx1);
  group_h_k<<<cdiv((long long)Bc * 512 * 32, 256), 256, 0, stream>>>(pts, pts, Bc, 4096, 3, co1, gidx1, 512, 32, 32, H0);
  const int M1 = Bc * 512 * 32;  // 524288
  run_layer(H0, bufY, H1, M1, sa1[0]);
  run_layer(H1, bufY, H0, M1, sa1[1]);
  run_layer(H0, bufY, H1, M1, sa1[2]);
  maxpool_k<<<cdiv((long long)Bc * 512 * 128, 256), 256, 0, stream>>>(bufY, fea1, (long long)Bc * 512, 32, 128);

  // ---------------- SA2: 512 -> 128, r=0.4, K=64, [131,128,128,256] --------
  fps_k<<<Bc, 256, 0, stream>>>(co1, 512, 128, idx2, co2);
  ball_query_k<<<cdiv((long long)Bc * 128, 256), 256, 0, stream>>>(co1, co2, Bc, 512, 128, 64, 0.16f, gidx2);
  group_h_k<<<cdiv((long long)Bc * 128 * 64, 256), 256, 0, stream>>>(co1, fea1, Bc, 512, 128, co2, gidx2, 128, 64, 160, H0);
  const int M2 = Bc * 128 * 64;  // 262144
  run_layer(H0, bufY, H1, M2, sa2[0]);
  run_layer(H1, bufY, H0, M2, sa2[1]);
  run_layer(H0, bufY, H1, M2, sa2[2]);
  maxpool_k<<<cdiv((long long)Bc * 128 * 256, 256), 256, 0, stream>>>(bufY, fea2, (long long)Bc * 128, 64, 256);

  // ---------------- SA3: 128 -> 64, global, [256,256,512,1024] -------------
  fps_k<<<Bc, 256, 0, stream>>>(co2, 128, 64, idx3, co3);
  gather_fea_h_k<<<cdiv((long long)Bc * 64 * 256, 256), 256, 0, stream>>>(fea2, idx3, Bc, 128, 64, 256, H0);
  const int M3 = Bc * 64;  // 2048
  run_layer(H0, bufY, H1, M3, sa3[0]);
  run_layer(H1, bufY, H0, M3, sa3[1]);
  run_layer(H0, bufY, H1, M3, sa3[2]);
  maxpool_k<<<cdiv((long long)Bc * 64 * 1024, 256), 256, 0, stream>>>(bufY, fea3, (long long)Bc * 64, 1, 1024);

  // ---------------- FP3: interp fea3 onto co2, concat fea2, [1280,256,256] -
  copy_cols_h_k<<<cdiv((long long)Bc * 128 * 256, 256), 256, 0, stream>>>(fea2, H0, (long long)Bc * 128, 256, 1280, 0);
  fp_interp_h_k<<<cdiv((long long)Bc * 128, 256), 256, 0, stream>>>(co2, co3, fea3, Bc, 128, 64, 1024, H0, 1280, 256);
  const int Mf3 = Bc * 128;  // 4096
  run_layer(H0, bufY, H1, Mf3, fp3[0]);
  run_layer(H1, bufY, H0, Mf3, fp3[1]);  // f2: f32 in bufY [32,128,256]

  // ---------------- FP2: interp f2 onto co1, concat fea1, [384,256,128] ----
  copy_cols_h_k<<<cdiv((long long)Bc * 512 * 128, 256), 256, 0, stream>>>(fea1, H1, (long long)Bc * 512, 128, 384, 0);
  fp_interp_h_k<<<cdiv((long long)Bc * 512, 256), 256, 0, stream>>>(co1, co2, bufY, Bc, 512, 128, 256, H1, 384, 128);
  const int Mf2 = Bc * 512;  // 16384
  run_layer(H1, bufY, H0, Mf2, fp2[0]);
  run_layer(H0, bufY, H1, Mf2, fp2[1]);  // f1: f32 in bufY [32,512,128]

  // ---------------- FP1: interp f1 onto pts, [128,128x4] -------------------
  fp_interp_h_k<<<cdiv((long long)Bc * 4096, 256), 256, 0, stream>>>(pts, co1, bufY, Bc, 4096, 512, 128, H0, 128, 0);
  const int Mf1 = Bc * 4096;  // 131072
  run_layer(H0, bufY, H1, Mf1, fp1[0]);
  run_layer(H1, bufY, H0, Mf1, fp1[1]);
  run_layer(H0, bufY, H1, Mf1, fp1[2]);
  run_layer(H1, bufY, H0, Mf1, fp1[3]);  // f0: f16 in H0

  // ---------------- head GEMM (N=2, padded to 16) + point-axis softmax -----
  run_gemm(H0, (float*)d_out, Mf1, head);
  softmax_points_k<<<dim3(Bc, 2), 256, 0, stream>>>((float*)d_out, 4096, 2);
}
